// TransformerIntentionLayer_65695819760379
// MI455X (gfx1250) — compile-verified, matrix operand order checked by reading, not run on hardware
//
#include <hip/hip_runtime.h>
#include <hip/hip_bf16.h>
#include <math.h>

typedef __attribute__((ext_vector_type(16))) _Float16 v16h;
typedef __attribute__((ext_vector_type(8)))  float    v8f;

#define M_TOK 2048   // T*K*N = 32*8*8
#define D_    256
#define DD_   65536  // D*D

// ---------------------------------------------------------------------------
// WMMA GEMM: C[M x N] (row stride ldc) =
//        act( A[M x K] @ W[K x N] + sb*bias[N] + add[M x N] )
// act: 0=none 1=relu 2=tanh 3=sigmoid.
// Block = 256 threads (8 waves) computing a 64(M) x 128(N) tile; waves are
// arranged 4(M) x 2(N), each wave owns a 16x64 sub-tile (4 accumulators).
// The B panel (32x128 per k-step) is cooperatively staged into LDS in
// WMMA-fragment order: coalesced b128 global loads + packed f32->f16 cvts in,
// one 32-byte LDS vector read (2 x ds_load_b128) per fragment out.
// ---------------------------------------------------------------------------
__global__ __launch_bounds__(256)
void gemm_wmma(const float* __restrict__ A, const float* __restrict__ W,
               const float* __restrict__ bias, const float* __restrict__ add,
               float* __restrict__ C, int M, int N, int K, int ldc,
               float sb, int act)
{
  __shared__ _Float16 bpanel[8 * 32 * 16];   // [jj 0..7][lane 0..31][e 0..15], 8KB

  const int tid  = threadIdx.x;
  const int wave = tid >> 5;
  const int lane = tid & 31;
  const int half = lane >> 4;
  const int r    = lane & 15;
  const int wm   = wave >> 1;                // 0..3 : M sub-block
  const int wn   = wave & 1;                 // 0..1 : N sub-block

  const int tm  = blockIdx.x * 64 + wm * 16;
  const int tn0 = blockIdx.y * 128;          // block column base
  const int tnw = tn0 + wn * 64;             // wave column base

  const float* arow = A + (size_t)(tm + r) * K;
  v8f acc[4] = {};

  for (int k0 = 0; k0 < K; k0 += 32) {
    __syncthreads();                         // protect previous iteration reads
    // ---- stage B panel rows k0..k0+31, cols tn0..tn0+127 in fragment order --
#pragma unroll
    for (int it = 0; it < 4; ++it) {
      const int idx4 = (it * 256 + tid) * 4; // element index within 32x128 panel
      const int k = idx4 >> 7;               // 0..31
      const int n = idx4 & 127;              // multiple of 4
      const float4 w4 = *(const float4*)(W + (size_t)(k0 + k) * N + tn0 + n);
      // inverse of the 16-bit fragment K mapping (ISA 7.12.2):
      const int hf = (k >> 3) & 1;                              // lane half
      const int i  = (k < 16) ? ((k & 7) >> 1) : (4 + ((k & 7) >> 1));
      const int e  = 2 * i + (k & 1);                           // element slot
      const int jj = n >> 4;                                    // 16-col subtile
      _Float16* dst = &bpanel[(jj * 32 + hf * 16 + (n & 15)) * 16 + e];
      dst[0 * 16] = (_Float16)w4.x;
      dst[1 * 16] = (_Float16)w4.y;
      dst[2 * 16] = (_Float16)w4.z;
      dst[3 * 16] = (_Float16)w4.w;
    }
    __syncthreads();

    // ---- A fragment: two contiguous 8-float runs -> vectorized b128 loads --
    __builtin_prefetch(arow + k0 + 64, 0, 1);
    const float4* apl = (const float4*)(arow + k0 + 8 * half);
    const float4* aph = (const float4*)(arow + k0 + 16 + 8 * half);
    const float4 x0 = apl[0], x1 = apl[1], x2 = aph[0], x3 = aph[1];
    v16h a;
    a[0]=(_Float16)x0.x; a[1]=(_Float16)x0.y; a[2]=(_Float16)x0.z; a[3]=(_Float16)x0.w;
    a[4]=(_Float16)x1.x; a[5]=(_Float16)x1.y; a[6]=(_Float16)x1.z; a[7]=(_Float16)x1.w;
    a[8]=(_Float16)x2.x; a[9]=(_Float16)x2.y; a[10]=(_Float16)x2.z; a[11]=(_Float16)x2.w;
    a[12]=(_Float16)x3.x; a[13]=(_Float16)x3.y; a[14]=(_Float16)x3.z; a[15]=(_Float16)x3.w;

    // ---- 4 WMMAs sharing the A fragment; B fragments straight from LDS -----
#pragma unroll
    for (int j = 0; j < 4; ++j) {
      const int jj = wn * 4 + j;
      const v16h b = *(const v16h*)&bpanel[(jj * 32 + lane) * 16];
      acc[j] = __builtin_amdgcn_wmma_f32_16x16x32_f16(false, a, false, b,
                                                      (short)0, acc[j], false, false);
    }
  }

#pragma unroll
  for (int j = 0; j < 4; ++j) {
#pragma unroll
    for (int v = 0; v < 8; ++v) {
      const int m = tm + v + 8 * half;
      const int n = tnw + 16 * j + r;
      float x = acc[j][v];
      if (bias) x += sb * bias[n];
      if (add)  x += add[(size_t)m * N + n];
      if (act == 1)      x = fmaxf(x, 0.0f);
      else if (act == 2) x = tanhf(x);
      else if (act == 3) x = 1.0f / (1.0f + __expf(-x));
      C[(size_t)m * ldc + n] = x;
    }
  }
}

// ---------------------------------------------------------------------------
// Fused hypernet:  qi[tok,e] = tanh( sum_{h,d} hmid[tok,h]*qint[tok,d]*W2[h,d*256+e]
//                                   + sum_d qint[tok,d]*b2[d*256+e] )
// One block = 16 tokens, 8 waves. hmid/qint tiles are staged into LDS with
// GLOBAL_LOAD_ASYNC_TO_LDS_B128 (ASYNCcnt), then each wave runs WMMA chunks of
// G = hmid16 @ W2[:, de:de+64] and contracts with qint[d] into an LDS
// accumulator via ds_add_f32.  W2 (64MB) stays L2-resident across 128 blocks.
// ---------------------------------------------------------------------------
__global__ void hypernet_qi(const float* __restrict__ hmid, const float* __restrict__ qint,
                            const float* __restrict__ W2,   const float* __restrict__ b2,
                            float* __restrict__ qi)
{
  __shared__ float    stageh[16 * 256];
  __shared__ float    qv[16 * 256];
  __shared__ _Float16 hA[16 * 256];
  __shared__ float    accs[16 * 256];

  const int tid  = threadIdx.x;
  const int tok0 = blockIdx.x * 16;

  // ---- async stage of hmid/qint tiles (16B per lane per op) ----
  const float* gh = hmid + (size_t)tok0 * 256;
  const float* gq = qint + (size_t)tok0 * 256;
#pragma unroll
  for (int it = 0; it < 4; ++it) {
    const int idx = (it * 256 + tid) * 4;
    const unsigned lh = (unsigned)(uintptr_t)&stageh[idx];
    const unsigned lq = (unsigned)(uintptr_t)&qv[idx];
    asm volatile("global_load_async_to_lds_b128 %0, %1, off"
                 :: "v"(lh), "v"(gh + idx) : "memory");
    asm volatile("global_load_async_to_lds_b128 %0, %1, off"
                 :: "v"(lq), "v"(gq + idx) : "memory");
  }
#if __has_builtin(__builtin_amdgcn_s_wait_asynccnt)
  __builtin_amdgcn_s_wait_asynccnt(0);
#else
  asm volatile("s_wait_asynccnt 0x0" ::: "memory");
#endif
  __syncthreads();
  for (int i = tid; i < 16 * 256; i += 256) {
    hA[i]   = (_Float16)stageh[i];
    accs[i] = 0.0f;
  }
  __syncthreads();

  const int wave = tid >> 5;
  const int lane = tid & 31;
  const int half = lane >> 4;
  const int r    = lane & 15;

  for (int c = wave; c < 1024; c += 8) {     // 1024 chunks of 64 over de=65536
    const int de0 = c * 64;
    const int d   = de0 >> 8;                // 64-chunk never crosses a d boundary
    const int e0  = de0 & 255;
    v8f acc[4] = {};
    for (int k0 = 0; k0 < 256; k0 += 32) {
      v16h a;
#pragma unroll
      for (int i = 0; i < 8; ++i) {
        const int kk = ((i < 4) ? 2 * i : 16 + 2 * (i - 4)) + 8 * half;
        a[2 * i]     = hA[r * 256 + k0 + kk];
        a[2 * i + 1] = hA[r * 256 + k0 + kk + 1];
      }
#pragma unroll
      for (int j = 0; j < 4; ++j) {
        v16h b;
#pragma unroll
        for (int i = 0; i < 8; ++i) {
          const int kk = ((i < 4) ? 2 * i : 16 + 2 * (i - 4)) + 8 * half;
          b[2 * i]     = (_Float16)W2[(size_t)(k0 + kk)     * DD_ + de0 + 16 * j + r];
          b[2 * i + 1] = (_Float16)W2[(size_t)(k0 + kk + 1) * DD_ + de0 + 16 * j + r];
        }
        acc[j] = __builtin_amdgcn_wmma_f32_16x16x32_f16(false, a, false, b,
                                                        (short)0, acc[j], false, false);
      }
    }
#pragma unroll
    for (int j = 0; j < 4; ++j) {
      const float bb = b2[de0 + 16 * j + r];
#pragma unroll
      for (int v = 0; v < 8; ++v) {
        const int tok = v + 8 * half;
        atomicAdd(&accs[tok * 256 + e0 + 16 * j + r],
                  qv[tok * 256 + d] * (acc[j][v] + bb));
      }
    }
  }
  __syncthreads();
  for (int i = tid; i < 16 * 256; i += 256)
    qi[(size_t)tok0 * 256 + i] = tanhf(accs[i]);
#if __has_builtin(__builtin_amdgcn_s_cluster_barrier)
  __builtin_amdgcn_s_cluster_barrier();    // NOP when ClusterID==0
#endif
}

// ---------------------------------------------------------------------------
// Tiny self-attention over N=8 (stages 1 and 3). One thread per (group,h,m).
// q,k: width Dq (256 or 512, 8 heads); v/out: width Dv=256 (head dim 32).
// ---------------------------------------------------------------------------
__global__ void attn_small(const float* __restrict__ q, const float* __restrict__ k,
                           const float* __restrict__ v, float* __restrict__ out,
                           int Dq, int Dv)
{
  const int idx = blockIdx.x * blockDim.x + threadIdx.x;
  const int m = idx & 7, h = (idx >> 3) & 7, g = idx >> 6;
  if (g >= 256) return;
  const int dqh = Dq >> 3;
  const float scale = rsqrtf((float)dqh);
  const float* qp = q + ((size_t)g * 8 + m) * Dq + h * dqh;

  float s[8], mx = -1e30f;
  for (int j = 0; j < 8; ++j) {
    const float* kp = k + ((size_t)g * 8 + j) * Dq + h * dqh;
    float d = 0.f;
    for (int c = 0; c < dqh; ++c) d += qp[c] * kp[c];
    s[j] = d * scale;
    mx = fmaxf(mx, s[j]);
  }
  float sum = 0.f;
  for (int j = 0; j < 8; ++j) { s[j] = __expf(s[j] - mx); sum += s[j]; }
  const float inv = 1.0f / sum;

  float o[32];
  for (int c = 0; c < 32; ++c) o[c] = 0.f;
  for (int j = 0; j < 8; ++j) {
    const float* vp = v + ((size_t)g * 8 + j) * Dv + h * 32;
    const float w = s[j] * inv;
    for (int c = 0; c < 32; ++c) o[c] += w * vp[c];
  }
  float* op = out + ((size_t)g * 8 + m) * Dv + h * 32;
  for (int c = 0; c < 32; ++c) op[c] = o[c];
}

// ---------------------------------------------------------------------------
// Cross attention: q[32,64,256], k/v[2048,256] (broadcast over t), 8 heads,
// head dim 32, online softmax over 2048 keys. One thread per (t,h,m).
// ---------------------------------------------------------------------------
__global__ void attn_cross(const float* __restrict__ q, const float* __restrict__ k,
                           const float* __restrict__ v, float* __restrict__ out)
{
  const int idx = blockIdx.x * blockDim.x + threadIdx.x;
  const int m = idx & 63, h = (idx >> 6) & 7, t = idx >> 9;
  if (t >= 32) return;
  const float* qp = q + ((size_t)t * 64 + m) * 256 + h * 32;
  float qr[32];
  for (int c = 0; c < 32; ++c) qr[c] = qp[c] * 0.1767766953f;  // 1/sqrt(32)

  float mx = -1e30f, sum = 0.f, o[32];
  for (int c = 0; c < 32; ++c) o[c] = 0.f;
  for (int j = 0; j < 2048; ++j) {
    const float* kp = k + (size_t)j * 256 + h * 32;
    __builtin_prefetch(kp + 512, 0, 1);
    float d = 0.f;
    for (int c = 0; c < 32; ++c) d += qr[c] * kp[c];
    const float nm = fmaxf(mx, d);
    const float corr = __expf(mx - nm);
    const float w = __expf(d - nm);
    sum = sum * corr + w;
    const float* vp = v + (size_t)j * 256 + h * 32;
    for (int c = 0; c < 32; ++c) o[c] = o[c] * corr + w * vp[c];
    mx = nm;
  }
  const float inv = 1.0f / sum;
  float* op = out + ((size_t)t * 64 + m) * 256 + h * 32;
  for (int c = 0; c < 32; ++c) op[c] = o[c] * inv;
}

// ------------------------- elementwise / layout kernels --------------------
__global__ void layernorm_k(const float* __restrict__ x, const float* __restrict__ g,
                            const float* __restrict__ b, float* __restrict__ y)
{
  __shared__ float red[256];
  const int t = blockIdx.x, c = threadIdx.x;
  const float v = x[(size_t)t * 256 + c];
  red[c] = v; __syncthreads();
  for (int s = 128; s > 0; s >>= 1) { if (c < s) red[c] += red[c + s]; __syncthreads(); }
  const float mean = red[0] * (1.0f / 256.0f);
  __syncthreads();
  const float dv = v - mean;
  red[c] = dv * dv; __syncthreads();
  for (int s = 128; s > 0; s >>= 1) { if (c < s) red[c] += red[c + s]; __syncthreads(); }
  const float var = red[0] * (1.0f / 256.0f);
  y[(size_t)t * 256 + c] = dv * rsqrtf(var + 1e-6f) * g[c] + b[c];
}

__global__ void ew_add_inplace(float* a, const float* __restrict__ b, int n)
{ int i = blockIdx.x * blockDim.x + threadIdx.x; if (i < n) a[i] += b[i]; }

__global__ void ew_mul(const float* __restrict__ a, const float* __restrict__ b,
                       float* __restrict__ o, int n)
{ int i = blockIdx.x * blockDim.x + threadIdx.x; if (i < n) o[i] = a[i] * b[i]; }

__global__ void ew_gate(const float* __restrict__ x, const float* __restrict__ z,
                        const float* __restrict__ h, float* __restrict__ o, int n)
{
  int i = blockIdx.x * blockDim.x + threadIdx.x;
  if (i < n) { float zz = z[i]; o[i] = (1.0f - zz) * x[i] + zz * h[i]; }
}

// swap axes 1,2 of (32,8,8,256):  out[t,b,a,c] = in[t,a,b,c]
__global__ void swap12(const float* __restrict__ in, float* __restrict__ out)
{
  int i = blockIdx.x * blockDim.x + threadIdx.x;
  if (i >= 32 * 8 * 8 * 256) return;
  const int c = i & 255, b = (i >> 8) & 7, a = (i >> 11) & 7, t = i >> 14;
  out[((((size_t)t * 8 + b) * 8 + a) << 8) + c] = in[i];
}

// src[2048,256] -> dst[2048,512] columns 256..511
__global__ void concat_copy(const float* __restrict__ src, float* __restrict__ dst)
{
  int i = blockIdx.x * blockDim.x + threadIdx.x;
  if (i >= 2048 * 256) return;
  const int m = i >> 8, c = i & 255;
  dst[(size_t)m * 512 + 256 + c] = src[i];
}

// ------------------------------- host side ---------------------------------
static inline void gemm(hipStream_t st, const float* A, const float* W,
                        const float* bias, const float* add, float* C,
                        int M, int N, int K, int ldc, float sb, int act)
{
  gemm_wmma<<<dim3(M / 64, N / 128), 256, 0, st>>>(A, W, bias, add, C, M, N, K, ldc, sb, act);
}

static void run_gating(hipStream_t st, const float* x, const float* y,
                       const float* Wr, const float* Ur, const float* Wz,
                       const float* Uz, const float* Wg, const float* Ug,
                       const float* bg, float* t0, float* r, float* z,
                       float* rx, float* hh, float* out)
{
  const int n = M_TOK * D_;
  gemm(st, y, Wr, nullptr, nullptr, t0, M_TOK, D_, D_, D_, 1.f, 0);
  gemm(st, x, Ur, nullptr, t0, r, M_TOK, D_, D_, D_, 1.f, 3);          // r = sigmoid
  gemm(st, y, Wz, nullptr, nullptr, t0, M_TOK, D_, D_, D_, 1.f, 0);
  gemm(st, x, Uz, bg, t0, z, M_TOK, D_, D_, D_, -1.f, 3);              // z = sigmoid(..-bg)
  gemm(st, y, Wg, nullptr, nullptr, t0, M_TOK, D_, D_, D_, 1.f, 0);
  ew_mul<<<dim3((n + 255) / 256), 256, 0, st>>>(r, x, rx, n);
  gemm(st, rx, Ug, nullptr, t0, hh, M_TOK, D_, D_, D_, 1.f, 2);        // h = tanh
  ew_gate<<<dim3((n + 255) / 256), 256, 0, st>>>(x, z, hh, out, n);
}

static void run_ffn(hipStream_t st, const float* xin, const float* W1, const float* b1,
                    const float* W2, const float* b2, float* mid, float* out)
{
  gemm(st, xin, W1, b1, nullptr, mid, M_TOK, 512, 256, 512, 1.f, 1);   // relu
  gemm(st, mid, W2, b2, nullptr, out, M_TOK, 256, 512, 256, 1.f, 0);
}

#define IN(i) ((const float*)d_in[(i)])

extern "C" void kernel_launch(void* const* d_in, const int* in_sizes, int n_in,
                              void* d_out, int out_size, void* d_ws, size_t ws_size,
                              hipStream_t stream)
{
  (void)in_sizes; (void)n_in; (void)out_size; (void)ws_size;
  const float* adv  = IN(0);
  const float* kint = IN(1);
  const float* qint = IN(2);
  // params flattened in dict order: 3..94
  float* out_main = (float*)d_out;
  float* out_sca  = out_main + (size_t)M_TOK * D_;

  const size_t SLOT = (size_t)M_TOK * 512;        // 4MB slots (fits 512-wide)
  float* S = (float*)d_ws;
  auto s = [&](int i) { return S + (size_t)i * SLOT; };
  const int n1 = M_TOK * D_;
  const dim3 EWG((n1 + 255) / 256), EWB(256);

  // ------------------ stage 1: self attention + hypernet --------------------
  gemm(stream, adv,  IN(3),  IN(4),  nullptr, s(1), M_TOK, D_, D_, D_, 1.f, 0);   // q_s
  gemm(stream, adv,  IN(5),  IN(6),  nullptr, s(2), M_TOK, D_, D_, D_, 1.f, 0);   // k_s
  gemm(stream, adv,  IN(7),  IN(8),  nullptr, s(3), M_TOK, D_, D_, D_, 1.f, 0);   // v_s
  gemm(stream, qint, IN(21), IN(22), nullptr, s(4), M_TOK, D_, D_, D_, 1.f, 1);   // hmid=relu
  hypernet_qi<<<dim3(128), 256, 0, stream>>>(s(4), qint, IN(23), IN(24), s(0));   // qi
  ew_add_inplace<<<EWG, EWB, 0, stream>>>(s(1), s(0), n1);                        // q_s+qi
  ew_add_inplace<<<EWG, EWB, 0, stream>>>(s(2), s(0), n1);                        // k_s+qi
  attn_small<<<dim3(64), 256, 0, stream>>>(s(1), s(2), s(3), s(5), 256, 256);
  gemm(stream, s(5), IN(25), IN(26), nullptr, s(6), M_TOK, D_, D_, D_, 1.f, 0);   // sa
  run_gating(stream, adv, s(6), IN(31), IN(32), IN(33), IN(34), IN(35), IN(36), IN(37),
             s(9), s(10), s(11), s(12), s(13), s(7));                             // g1
  layernorm_k<<<dim3(M_TOK), 256, 0, stream>>>(s(7), IN(85), IN(86), s(9));
  run_ffn(stream, s(9), IN(73), IN(74), IN(75), IN(76), s(14), s(9));
  run_gating(stream, s(7), s(9), IN(38), IN(39), IN(40), IN(41), IN(42), IN(43), IN(44),
             s(10), s(11), s(12), s(13), s(1), s(8));                             // tx

  // ------------------ stage 2: cross attention ------------------------------
  layernorm_k<<<dim3(M_TOK), 256, 0, stream>>>(s(8), IN(87), IN(88), s(9));
  swap12<<<EWG, EWB, 0, stream>>>(s(9), s(15));                                   // q_cross0
  swap12<<<EWG, EWB, 0, stream>>>(s(0), s(16));                                   // qi perm
  gemm(stream, s(15), IN(9),  IN(10), s(16),  s(17), M_TOK, D_, D_, D_, 1.f, 0);  // q_cross
  gemm(stream, kint,  IN(11), IN(12), nullptr, s(18), M_TOK, D_, D_, D_, 1.f, 0); // k_c
  gemm(stream, kint,  IN(13), IN(14), nullptr, s(19), M_TOK, D_, D_, D_, 1.f, 0); // v_c
  attn_cross<<<dim3(64), 256, 0, stream>>>(s(17), s(18), s(19), s(5));
  gemm(stream, s(5), IN(27), IN(28), nullptr, s(6), M_TOK, D_, D_, D_, 1.f, 0);   // ca
  run_gating(stream, s(15), s(6), IN(45), IN(46), IN(47), IN(48), IN(49), IN(50), IN(51),
             s(9), s(10), s(11), s(12), s(13), s(7));                             // gc
  layernorm_k<<<dim3(M_TOK), 256, 0, stream>>>(s(7), IN(89), IN(90), s(9));
  run_ffn(stream, s(9), IN(77), IN(78), IN(79), IN(80), s(14), s(9));
  run_gating(stream, s(7), s(9), IN(52), IN(53), IN(54), IN(55), IN(56), IN(57), IN(58),
             s(10), s(11), s(12), s(13), s(17), s(8));                            // tc

  // ------------------ stage 3: concat-channel self attention ----------------
  layernorm_k<<<dim3(M_TOK), 256, 0, stream>>>(s(8), IN(91), IN(92), s(9));
  swap12<<<EWG, EWB, 0, stream>>>(s(9), s(20));                                   // q_sc0
  gemm(stream, s(20), IN(15), IN(16), nullptr, s(21), M_TOK, D_, D_, 512, 1.f, 0);
  concat_copy<<<EWG, EWB, 0, stream>>>(s(0), s(21));                              // q_sc
  gemm(stream, s(20), IN(17), IN(18), nullptr, s(22), M_TOK, D_, D_, 512, 1.f, 0);
  concat_copy<<<EWG, EWB, 0, stream>>>(s(0), s(22));                              // k_sc
  gemm(stream, s(20), IN(19), IN(20), nullptr, s(23), M_TOK, D_, D_, D_, 1.f, 0); // v_sc
  attn_small<<<dim3(64), 256, 0, stream>>>(s(21), s(22), s(23), s(5), 512, 256);
  gemm(stream, s(5), IN(29), IN(30), nullptr, out_sca, M_TOK, D_, D_, D_, 1.f, 0);// sca (output 2)
  run_gating(stream, s(20), out_sca, IN(59), IN(60), IN(61), IN(62), IN(63), IN(64), IN(65),
             s(9), s(10), s(11), s(12), s(13), s(6));                             // gsc
  layernorm_k<<<dim3(M_TOK), 256, 0, stream>>>(s(6), IN(93), IN(94), s(10));
  run_ffn(stream, s(10), IN(81), IN(82), IN(83), IN(84), s(14), s(10));
  run_gating(stream, s(6), s(10), IN(66), IN(67), IN(68), IN(69), IN(70), IN(71), IN(72),
             s(9), s(11), s(12), s(13), s(15), s(7));                             // tsc
  layernorm_k<<<dim3(M_TOK), 256, 0, stream>>>(s(7), IN(91), IN(92), out_main);   // output 1
}
